// IRS_38714835206344
// MI455X (gfx1250) — matrix operand order repaired and checked by reading, not compiled, not run on hardware
//
#include <hip/hip_runtime.h>
#include <hip/hip_bf16.h>
#include <math.h>

// ---------------- problem constants ----------------
#define BB   2
#define CC   128          // DIM
#define HC   64           // DIM/2
#define HH   256
#define WW   256
#define HWSZ (HH * WW)    // 65536
#define QKVO (3 * CC)     // 384

typedef __attribute__((ext_vector_type(16))) __bf16 v16bf;
typedef __attribute__((ext_vector_type(8)))  float  v8f;
typedef unsigned int u32x4 __attribute__((ext_vector_type(4)));
typedef int          i32x4 __attribute__((ext_vector_type(4)));
typedef int          i32x8 __attribute__((ext_vector_type(8)));

// ---- gfx1250 feature availability (device pass only; host parses fallback) ----
#if defined(__gfx1250__) && __has_builtin(__builtin_amdgcn_tensor_load_to_lds)
#define HAVE_TDM 1
#else
#define HAVE_TDM 0
#endif
#if defined(__gfx1250__) && __has_builtin(__builtin_amdgcn_global_load_async_to_lds_b128)
#define HAVE_ASYNC 1
#else
#define HAVE_ASYNC 0
#endif

// WMMA operand index helpers (CDNA5 ISA VGPR layouts, wave32)
// A matrix 16x32 bf16: lane holds row m = lane&15; element e -> K index:
__device__ __forceinline__ int a_k(int e, int lane) {
    return (e & 7) + ((e >> 3) << 4) + ((lane >> 4) << 3);
}
// B matrix 32x16 bf16: lane holds col n = lane&15; element e -> K index (same as A):
__device__ __forceinline__ int b_k(int e, int lane) {
    return (e & 7) + ((e >> 3) << 4) + ((lane >> 4) << 3);
}
// C/D 16x16 f32: vgpr r -> row m = r + ((lane>>4)<<3), col n = lane&15.

// LDS byte offset of a __shared__ object (addrspacecast flat->local drops aperture)
__device__ __forceinline__ unsigned lds_offset(const void* p) {
#if defined(__AMDGCN__)
    return (unsigned)(size_t)(__attribute__((address_space(3))) const void*)p;
#else
    (void)p; return 0u;
#endif
}

__device__ __forceinline__ void wait_tensorcnt0() {
#if defined(__gfx1250__)
#if __has_builtin(__builtin_amdgcn_s_wait_tensorcnt)
    __builtin_amdgcn_s_wait_tensorcnt((short)0);
#else
    asm volatile("s_wait_tensorcnt 0x0" ::: "memory");
#endif
    asm volatile("" ::: "memory");
#endif
}

__device__ __forceinline__ void wait_asynccnt0() {
#if defined(__gfx1250__)
#if __has_builtin(__builtin_amdgcn_s_wait_asynccnt)
    __builtin_amdgcn_s_wait_asynccnt((short)0);
#else
    asm volatile("s_wait_asynccnt 0x0" ::: "memory");
#endif
    asm volatile("" ::: "memory");
#endif
}

#if HAVE_TDM
// Build a 2D Tensor-DMA descriptor (CDNA5 ISA) and
// issue TENSOR_LOAD_TO_LDS: copies rows x cols bf16 elements (row stride = cols,
// i.e. a contiguous block) from global memory into LDS at lds_byte.
__device__ __forceinline__ void tdm_load_rows_bf16(const __bf16* gsrc, unsigned lds_byte,
                                                   int rows, int cols) {
    unsigned long long ga = (unsigned long long)(size_t)gsrc;
    u32x4 g0 = { 1u,                                   // count=1, no gather, user mode
                 lds_byte,                             // lds_addr (bytes)
                 (unsigned)ga,                         // global_addr[31:0]
                 (unsigned)((ga >> 32) & 0x01ffffffu)  // global_addr[56:32]
                     | (2u << 30) };                   // type = 2 ("image")
    i32x8 g1 = { (int)(1u << 16),                      // data_size=1 (2 bytes/elem)
                 (int)((unsigned)cols << 16),          // tensor_dim0[15:0]
                 (int)((unsigned)rows << 16),          // tensor_dim1[15:0]
                 (int)((unsigned)cols << 16),          // tile_dim0
                 rows,                                 // tile_dim1 ([15:0]); tile_dim2=0
                 cols,                                 // tensor_dim0_stride[31:0]
                 0, 0 };                               // stride hi / dim1_stride = 0
    i32x4 z4 = { 0, 0, 0, 0 };
#if __clang_major__ >= 23
    i32x8 z8 = { 0, 0, 0, 0, 0, 0, 0, 0 };
    __builtin_amdgcn_tensor_load_to_lds(g0, g1, z4, z4, z8, 0);
#else
    __builtin_amdgcn_tensor_load_to_lds(g0, g1, z4, z4, 0);
#endif
}
#endif

// =====================================================================
// 1) qkv = 1x1 conv: GEMM  Y[o, p] = sum_c W[o,c] * X[b,c,p] + bias[o]
// =====================================================================
__global__ void k_qkv(const float* __restrict__ x, const float* __restrict__ w,
                      const float* __restrict__ bias,
                      float* __restrict__ qb, float* __restrict__ kb, float* __restrict__ vb) {
    const int lane  = threadIdx.x;               // block = 32 (one wave)
    const int ptile = blockIdx.x;                // B*HW/16 tiles
    const int obase = blockIdx.y * 16;           // 0..368
    const int b     = ptile / (HWSZ / 16);
    const int pbase = (ptile % (HWSZ / 16)) * 16;

    const int am = obase + (lane & 15);
    const int bn = pbase + (lane & 15);
    const size_t xb = (size_t)b * CC * HWSZ;

    v8f acc = {};
    for (int kk = 0; kk < CC; kk += 32) {
        v16bf a, bm;
#pragma unroll
        for (int e = 0; e < 16; ++e) {
            a[e]  = (__bf16)w[am * CC + kk + a_k(e, lane)];
            bm[e] = (__bf16)x[xb + (size_t)(kk + b_k(e, lane)) * HWSZ + bn];
        }
        acc = __builtin_amdgcn_wmma_f32_16x16x32_bf16(false, a, false, bm,
                                                      (short)0, acc, false, false);
    }

    float* dst; int oloc;
    if (obase < CC)          { dst = qb; oloc = obase; }
    else if (obase < 2 * CC) { dst = kb; oloc = obase - CC; }
    else                     { dst = vb; oloc = obase - 2 * CC; }

    const int n = pbase + (lane & 15);
#pragma unroll
    for (int r = 0; r < 8; ++r) {
        const int mo = r + ((lane >> 4) << 3);
        dst[(size_t)b * CC * HWSZ + (size_t)(oloc + mo) * HWSZ + n] =
            acc[r] + bias[obase + mo];
    }
}

// =====================================================================
// 2) depthwise 3x3, SAME padding  (q and k paths)
// =====================================================================
__global__ void k_dw3x3(const float* __restrict__ in, const float* __restrict__ w,
                        const float* __restrict__ bias, float* __restrict__ out) {
    const size_t idx = (size_t)blockIdx.x * 256 + threadIdx.x;   // B*CC*HW exact
    const int wp = idx % WW;
    const int hp = (idx / WW) % HH;
    const int c  = (idx / HWSZ) % CC;
    const int b  = (int)(idx / ((size_t)CC * HWSZ));
    const float* src = in + ((size_t)b * CC + c) * HWSZ;
    float acc = bias[c];
#pragma unroll
    for (int dy = -1; dy <= 1; ++dy)
#pragma unroll
        for (int dx = -1; dx <= 1; ++dx) {
            const int y = hp + dy, xq = wp + dx;
            if (y >= 0 && y < HH && xq >= 0 && xq < WW)
                acc += w[c * 9 + (dy + 1) * 3 + (dx + 1)] * src[(size_t)y * WW + xq];
        }
    out[idx] = acc;
}

// =====================================================================
// 3) MDC stage 1: grouped conv (2 in-ch -> 1 out-ch per group), 3x3 SAME
// =====================================================================
__global__ void k_mdc1(const float* __restrict__ v, const float* __restrict__ w,
                       const float* __restrict__ bias, float* __restrict__ x1) {
    const size_t idx = (size_t)blockIdx.x * 256 + threadIdx.x;   // B*HC*HW exact
    const int wp = idx % WW;
    const int hp = (idx / WW) % HH;
    const int g  = (idx / HWSZ) % HC;
    const int b  = (int)(idx / ((size_t)HC * HWSZ));
    float acc = bias[g];
#pragma unroll
    for (int i = 0; i < 2; ++i) {
        const float* src = v + ((size_t)b * CC + (2 * g + i)) * HWSZ;
#pragma unroll
        for (int dy = -1; dy <= 1; ++dy)
#pragma unroll
            for (int dx = -1; dx <= 1; ++dx) {
                const int y = hp + dy, xq = wp + dx;
                if (y >= 0 && y < HH && xq >= 0 && xq < WW)
                    acc += w[g * 18 + i * 9 + (dy + 1) * 3 + (dx + 1)] *
                           src[(size_t)y * WW + xq];
            }
    }
    x1[idx] = acc;
}

// =====================================================================
// 4) MDC stage 2 + sigmoid gate; writes v transposed+bf16: vT[bc][w][h]
// =====================================================================
__global__ void k_gate(const float* __restrict__ v, const float* __restrict__ x1,
                       const float* __restrict__ w2, const float* __restrict__ b2,
                       __bf16* __restrict__ vT) {
    const size_t idx = (size_t)blockIdx.x * 256 + threadIdx.x;   // B*CC*HW exact
    const int wp = idx % WW;
    const int hp = (idx / WW) % HH;
    const int c  = (idx / HWSZ) % CC;
    const int b  = (int)(idx / ((size_t)CC * HWSZ));

    float m;
    if (c < HC) {
        m = x1[((size_t)b * HC + c) * HWSZ + (size_t)hp * WW + wp];
    } else {
        const int g = c - HC;
        const float* src = x1 + ((size_t)b * HC + g) * HWSZ;
        float acc = b2[g];
#pragma unroll
        for (int dy = -1; dy <= 1; ++dy)
#pragma unroll
            for (int dx = -1; dx <= 1; ++dx) {
                const int y = hp + dy, xq = wp + dx;
                if (y >= 0 && y < HH && xq >= 0 && xq < WW)
                    acc += w2[g * 9 + (dy + 1) * 3 + (dx + 1)] * src[(size_t)y * WW + xq];
            }
        m = acc;
    }
    const float s  = 1.0f / (1.0f + expf(-v[idx]));
    const float vg = s * m;
    vT[((size_t)b * CC + c) * HWSZ + (size_t)wp * HH + hp] = (__bf16)vg;
}

// =====================================================================
// 5) row L2 normalize over W, emit bf16 (row-major) for WMMA operands
// =====================================================================
__global__ void k_rownorm(const float* __restrict__ in, __bf16* __restrict__ out) {
    __shared__ float red[WW];
    const size_t row = blockIdx.x;               // B*CC*HH rows
    const int t = threadIdx.x;                   // 256 = W
    const float x = in[row * WW + t];
    red[t] = x * x;
    __syncthreads();
    for (int s = WW / 2; s > 0; s >>= 1) {
        if (t < s) red[t] += red[t + s];
        __syncthreads();
    }
    const float nrm = fmaxf(sqrtf(red[0]), 1e-12f);
    out[row * WW + t] = (__bf16)(x / nrm);
}

// =====================================================================
// 6) scores[bc][h][g] = (q1[h,:] . k1[g,:]) * temperature[c]   (K = W = 256)
//    Operand tiles staged into LDS by the Tensor Data Mover (TENSORcnt).
// =====================================================================
__global__ void k_scores(const __bf16* __restrict__ q1, const __bf16* __restrict__ k1,
                         const float* __restrict__ temp, float* __restrict__ sc) {
    __shared__ __attribute__((aligned(16))) __bf16 tA[16 * WW];   // 8KB
    __shared__ __attribute__((aligned(16))) __bf16 tB[16 * WW];   // 8KB
    const int lane = threadIdx.x;                // one wave
    const int bc   = blockIdx.y;                 // B*CC
    const int ti   = blockIdx.x & 15;            // h tile
    const int tj   = blockIdx.x >> 4;            // g tile
    const __bf16* aSrc = q1 + (size_t)bc * HWSZ + (size_t)(ti * 16) * WW;  // 16 full rows
    const __bf16* bSrc = k1 + (size_t)bc * HWSZ + (size_t)(tj * 16) * WW;  // contiguous 8KB

#if HAVE_TDM
    tdm_load_rows_bf16(aSrc, lds_offset(tA), 16, WW);
    tdm_load_rows_bf16(bSrc, lds_offset(tB), 16, WW);
    wait_tensorcnt0();
#else
    for (int i = lane; i < 16 * WW / 8; i += 32) {
        ((float4*)tA)[i] = ((const float4*)aSrc)[i];
        ((float4*)tB)[i] = ((const float4*)bSrc)[i];
    }
#endif

    const int ar = (lane & 15) * WW;             // this lane's A row (in LDS)
    const int br = (lane & 15) * WW;             // this lane's B row (in LDS)
    v8f acc = {};
    for (int kk = 0; kk < WW; kk += 32) {
        v16bf a, bm;
#pragma unroll
        for (int e = 0; e < 16; ++e) {
            a[e]  = tA[ar + kk + a_k(e, lane)];
            bm[e] = tB[br + kk + b_k(e, lane)];
        }
        acc = __builtin_amdgcn_wmma_f32_16x16x32_bf16(false, a, false, bm,
                                                      (short)0, acc, false, false);
    }
    const float ts = temp[bc % CC];
    float* dst = sc + (size_t)bc * HH * HH;
    const int n = tj * 16 + (lane & 15);
#pragma unroll
    for (int r = 0; r < 8; ++r) {
        const int m = ti * 16 + r + ((lane >> 4) << 3);
        dst[(size_t)m * HH + n] = acc[r] * ts;
    }
}

// =====================================================================
// 7) softmax over last dim (256), emit bf16 attn
// =====================================================================
__global__ void k_softmax(const float* __restrict__ sc, __bf16* __restrict__ attn) {
    __shared__ float red[HH];
    const size_t row = blockIdx.x;               // B*CC*HH rows
    const int t = threadIdx.x;                   // 256
    const float v = sc[row * HH + t];
    red[t] = v;
    __syncthreads();
    for (int s = HH / 2; s > 0; s >>= 1) {
        if (t < s) red[t] = fmaxf(red[t], red[t + s]);
        __syncthreads();
    }
    const float mx = red[0];
    __syncthreads();
    const float e = expf(v - mx);
    red[t] = e;
    __syncthreads();
    for (int s = HH / 2; s > 0; s >>= 1) {
        if (t < s) red[t] += red[t + s];
        __syncthreads();
    }
    attn[row * HH + t] = (__bf16)(e / red[0]);
}

// =====================================================================
// 8) out[bc][h][w] = sum_g attn[h,g] * v[g,w]
//    Operand tiles staged into LDS via async global->LDS B128 (ASYNCcnt).
// =====================================================================
__global__ void k_av(const __bf16* __restrict__ attn, const __bf16* __restrict__ vT,
                     __bf16* __restrict__ outbf) {
    __shared__ __attribute__((aligned(16))) __bf16 tA[16 * HH];   // 8KB
    __shared__ __attribute__((aligned(16))) __bf16 tB[16 * HH];   // 8KB
    const int lane = threadIdx.x;
    const int bc   = blockIdx.y;
    const int ti   = blockIdx.x & 15;            // h tile
    const int tj   = blockIdx.x >> 4;            // w tile
    const char* gA = (const char*)(attn + (size_t)bc * HH * HH + (size_t)(ti * 16) * HH);
    const char* gB = (const char*)(vT   + (size_t)bc * HWSZ   + (size_t)(tj * 16) * HH);

#if HAVE_ASYNC
#pragma unroll
    for (int i = 0; i < 16; ++i) {               // 16 wave-wide issues x 512B = 8KB/tile
        const unsigned off = (unsigned)i * 512u + (unsigned)lane * 16u;
        __builtin_amdgcn_global_load_async_to_lds_b128(
            (__attribute__((address_space(1))) i32x4*)(size_t)(gA + off),
            (__attribute__((address_space(3))) i32x4*)
                (__attribute__((address_space(3))) void*)((char*)tA + off),
            0, 0);
        __builtin_amdgcn_global_load_async_to_lds_b128(
            (__attribute__((address_space(1))) i32x4*)(size_t)(gB + off),
            (__attribute__((address_space(3))) i32x4*)
                (__attribute__((address_space(3))) void*)((char*)tB + off),
            0, 0);
    }
    wait_asynccnt0();
#else
    for (int i = lane; i < 16 * HH / 8; i += 32) {
        ((float4*)tA)[i] = ((const float4*)gA)[i];
        ((float4*)tB)[i] = ((const float4*)gB)[i];
    }
#endif

    const int ar = (lane & 15) * HH;
    const int br = (lane & 15) * HH;
    v8f acc = {};
    for (int kk = 0; kk < HH; kk += 32) {
        v16bf a, bm;
#pragma unroll
        for (int e = 0; e < 16; ++e) {
            a[e]  = tA[ar + kk + a_k(e, lane)];
            bm[e] = tB[br + kk + b_k(e, lane)];
        }
        acc = __builtin_amdgcn_wmma_f32_16x16x32_bf16(false, a, false, bm,
                                                      (short)0, acc, false, false);
    }
    __bf16* dst = outbf + (size_t)bc * HWSZ;
    const int n = tj * 16 + (lane & 15);
#pragma unroll
    for (int r = 0; r < 8; ++r) {
        const int m = ti * 16 + r + ((lane >> 4) << 3);
        dst[(size_t)m * WW + n] = (__bf16)acc[r];
    }
}

// =====================================================================
// 9) output projection: Y[o,p] = sum_c Wpo[o,c] * out[b,c,p] + bias[o]
// =====================================================================
__global__ void k_po(const __bf16* __restrict__ outbf, const float* __restrict__ w,
                     const float* __restrict__ bias, float* __restrict__ y) {
    const int lane  = threadIdx.x;
    const int ptile = blockIdx.x;
    const int obase = blockIdx.y * 16;           // 0..112
    const int b     = ptile / (HWSZ / 16);
    const int pbase = (ptile % (HWSZ / 16)) * 16;
    const int am = obase + (lane & 15);
    const int bn = pbase + (lane & 15);
    const size_t ob = (size_t)b * CC * HWSZ;

    v8f acc = {};
    for (int kk = 0; kk < CC; kk += 32) {
        v16bf a, bm;
#pragma unroll
        for (int e = 0; e < 16; ++e) {
            a[e]  = (__bf16)w[am * CC + kk + a_k(e, lane)];
            bm[e] = outbf[ob + (size_t)(kk + b_k(e, lane)) * HWSZ + bn];
        }
        acc = __builtin_amdgcn_wmma_f32_16x16x32_bf16(false, a, false, bm,
                                                      (short)0, acc, false, false);
    }
    const int n = pbase + (lane & 15);
#pragma unroll
    for (int r = 0; r < 8; ++r) {
        const int mo = r + ((lane >> 4) << 3);
        y[(size_t)b * CC * HWSZ + (size_t)(obase + mo) * HWSZ + n] =
            acc[r] + bias[obase + mo];
    }
}

// =====================================================================
// launcher
// =====================================================================
extern "C" void kernel_launch(void* const* d_in, const int* in_sizes, int n_in,
                              void* d_out, int out_size, void* d_ws, size_t ws_size,
                              hipStream_t stream) {
    (void)in_sizes; (void)n_in; (void)out_size; (void)ws_size;

    const float* x     = (const float*)d_in[0];
    const float* w_qkv = (const float*)d_in[1];
    const float* b_qkv = (const float*)d_in[2];
    const float* w_q   = (const float*)d_in[3];
    const float* b_q   = (const float*)d_in[4];
    const float* w_k   = (const float*)d_in[5];
    const float* b_k   = (const float*)d_in[6];
    const float* w_v1  = (const float*)d_in[7];
    const float* b_v1  = (const float*)d_in[8];
    const float* w_v2  = (const float*)d_in[9];
    const float* b_v2  = (const float*)d_in[10];
    const float* temp  = (const float*)d_in[11];
    const float* w_po  = (const float*)d_in[12];
    const float* b_po  = (const float*)d_in[13];
    float* y = (float*)d_out;

    // ---- workspace layout (bytes), aliased: needs ~448 MB ----
    char* ws = (char*)d_ws;
    const size_t F = (size_t)BB * CC * HWSZ;     // 16.7M elems per f32 tensor
    float*  qbuf   = (float*)(ws);                         // 64MB (freed after dw)
    float*  kbuf   = (float*)(ws + F * 4);                 // 64MB (freed after dw)
    float*  vbuf   = (float*)(ws + F * 8);                 // 64MB (freed after gate)
    float*  q2     = (float*)(ws + F * 12);                // 64MB
    float*  k2     = (float*)(ws + F * 16);                // 64MB
    float*  x1     = (float*)(ws + F * 20);                // 32MB (B*HC*HW)
    __bf16* q1bf   = (__bf16*)(ws + F * 22);               // 32MB
    __bf16* k1bf   = (__bf16*)(ws + F * 24);               // 32MB
    __bf16* vT     = (__bf16*)(ws + F * 26);               // 32MB
    float*  scores = qbuf;                                 // alias (64MB, B*C*H*H)
    __bf16* attn   = (__bf16*)kbuf;                        // alias (32MB)
    __bf16* outbf  = (__bf16*)vbuf;                        // alias (32MB)

    const dim3 wave(32);
    const int ptiles = BB * HWSZ / 16;                     // 8192

    // 1) qkv 1x1 conv (WMMA bf16)
    k_qkv<<<dim3(ptiles, QKVO / 16), wave, 0, stream>>>(x, w_qkv, b_qkv, qbuf, kbuf, vbuf);
    // 2) depthwise 3x3 on q, k
    k_dw3x3<<<(unsigned)(F / 256), 256, 0, stream>>>(qbuf, w_q, b_q, q2);
    k_dw3x3<<<(unsigned)(F / 256), 256, 0, stream>>>(kbuf, w_k, b_k, k2);
    // 3) MDC grouped conv + 4) depthwise + sigmoid gate -> vT (bf16, transposed)
    k_mdc1<<<(unsigned)(F / 512), 256, 0, stream>>>(vbuf, w_v1, b_v1, x1);
    k_gate<<<(unsigned)(F / 256), 256, 0, stream>>>(vbuf, x1, w_v2, b_v2, vT);
    // 5) L2 row-normalize -> bf16
    k_rownorm<<<(unsigned)(BB * CC * HH), WW, 0, stream>>>(q2, q1bf);
    k_rownorm<<<(unsigned)(BB * CC * HH), WW, 0, stream>>>(k2, k1bf);
    // 6) scores GEMMs (WMMA, TDM-staged LDS tiles)
    k_scores<<<dim3((HH / 16) * (HH / 16), BB * CC), wave, 0, stream>>>(q1bf, k1bf, temp, scores);
    // 7) softmax -> bf16 attn
    k_softmax<<<(unsigned)(BB * CC * HH), HH, 0, stream>>>(scores, attn);
    // 8) attn @ v GEMMs (WMMA, async global->LDS staged tiles)
    k_av<<<dim3((HH / 16) * (WW / 16), BB * CC), wave, 0, stream>>>(attn, vT, outbf);
    // 9) output projection (WMMA) + bias -> d_out (f32)
    k_po<<<dim3(ptiles, CC / 16), wave, 0, stream>>>(outbf, w_po, b_po, y);
}
